// Conv2D_BinaryLayer_18897856102731
// MI455X (gfx1250) — matrix operand order repaired and checked by reading, not compiled, not run on hardware
//
#include <hip/hip_runtime.h>

typedef __attribute__((ext_vector_type(16))) _Float16 v16h;
typedef __attribute__((ext_vector_type(8)))  _Float16 v8h;
typedef __attribute__((ext_vector_type(8)))  float    v8f;
typedef __attribute__((ext_vector_type(4)))  int      v4i;

#define BM 128
#define BN 128
#define BK 64          // K elements staged per chunk
#define KCHUNKS 18     // 1152 / 64

// ---- async-to-LDS availability (gfx1250 path) ----
#if __has_builtin(__builtin_amdgcn_global_load_async_to_lds_b128) && \
    __has_builtin(__builtin_amdgcn_s_wait_asynccnt)
#define HAVE_ASYNC 1
#else
#define HAVE_ASYNC 0
#endif

// ======================= pre-kernel: binarize + transpose B =======================
// wk [k=1152][n=256] fp32  ->  wsB [n=256][k=1152] f16 in {-1,+1}
__global__ __launch_bounds__(256) void binarize_B(const float* __restrict__ wk,
                                                  _Float16* __restrict__ wsB) {
    const int idx = blockIdx.x * 256 + threadIdx.x;   // 294912 elements
    const int k = idx >> 8;
    const int n = idx & 255;
    const float v = wk[idx];
    wsB[(size_t)n * 1152 + k] = (v > 0.0f) ? (_Float16)1.0f : (_Float16)-1.0f;
}

#if HAVE_ASYNC
// =============== async double-buffered WMMA kernel (A raw f32 in LDS) ===============
#define SA 68                      // f32 per A row (272B stride)
#define SB 72                      // f16 per B row (144B stride)
#define ABYTES (BM * SA * 4)       // 34816
#define BBYTES (BN * SB * 2)       // 18432
#define SMEM_TOTAL (2 * (ABYTES + BBYTES))   // 106496

typedef __attribute__((address_space(1))) v4i gas_v4i;  // global int4
typedef __attribute__((address_space(3))) v4i las_v4i;  // LDS int4

__device__ __forceinline__ void async_b128(const void* g, void* l) {
    // generic->AS casts via integer detour (AS3 pointer = low 32 bits = LDS offset)
    gas_v4i* gp = (gas_v4i*)(uintptr_t)g;
    las_v4i* lp = (las_v4i*)(unsigned int)(uintptr_t)l;
    __builtin_amdgcn_global_load_async_to_lds_b128(gp, lp, 0, 0);
}

__global__ __launch_bounds__(256) void binconv_wmma_async(
    const float* __restrict__ x,       // [32,112,112,128] fp32
    const _Float16* __restrict__ wsB,  // [256][1152] f16 (+/-1)
    const float* __restrict__ bias,
    float* __restrict__ out)
{
    extern __shared__ __align__(16) char smem[];
    char* const Abase = smem;                    // 2 x f32 buffers
    char* const Bbase = smem + 2 * ABYTES;       // 2 x f16 buffers

    const int tid  = threadIdx.x;
    const int lane = tid & 31;
    const int w    = tid >> 5;
    const int mw   = w & 3;
    const int nh   = w >> 2;

    const int Mbase = blockIdx.x * BM;
    const int Nbase = blockIdx.y * BN;

    // staging assignment
    const int a_m  = tid & 127;
    const int a_kb = (tid >> 7) << 5;    // {0,32} f32 within chunk
    const int b_n  = tid & 127;
    const int b_kb = (tid >> 7) << 5;    // {0,32} f16 within chunk

    const int mg  = Mbase + a_m;
    const int img = mg / 12100;
    const int rem = mg - img * 12100;
    const int oy  = rem / 110;
    const int ox  = rem - oy * 110;

    float bcol[4];
#pragma unroll
    for (int sn = 0; sn < 4; ++sn)
        bcol[sn] = bias[Nbase + nh * 64 + sn * 16 + (lane & 15)];

    v8f acc[2][4];
#pragma unroll
    for (int i = 0; i < 2; ++i)
#pragma unroll
        for (int j = 0; j < 4; ++j) acc[i][j] = (v8f){};

    // dest swizzle: within each 32-f32 K group store order [K0-7][K16-23][K8-15][K24-31]
    const int perm[8] = {0, 1, 4, 5, 2, 3, 6, 7};   // 16B-block permutation

    // issue all async copies for chunk kt into buffer buf
    auto issue = [&](int kt, int buf) {
        const int plane = kt >> 1;
        const int c0    = (kt & 1) << 6;
        const int ky    = plane / 3;
        const int kx    = plane - ky * 3;
        const char* gA = (const char*)(x +
            ((((img * 112) + (oy + ky)) * 112 + (ox + kx)) * 128 + c0 + a_kb));
        char* lA = Abase + buf * ABYTES + a_m * (SA * 4) + a_kb * 4;
#pragma unroll
        for (int i = 0; i < 8; ++i)                       // 8 x 16B = 128B
            async_b128(gA + 16 * i, lA + perm[i] * 16);
        const char* gB = (const char*)(wsB + (size_t)(Nbase + b_n) * 1152 + kt * BK + b_kb);
        char* lB = Bbase + buf * BBYTES + b_n * (SB * 2) + b_kb * 2;
#pragma unroll
        for (int i = 0; i < 4; ++i)                       // 4 x 16B = 64B
            async_b128(gB + 16 * i, lB + 16 * i);
    };

    issue(0, 0);

    const int koff  = (lane >> 4) << 4;       // 16-element group select (f32 / f16)
    const int mrow0 = mw * 32 + (lane & 15);
    const int nrow0 = nh * 64 + (lane & 15);

    for (int kt = 0; kt < KCHUNKS; ++kt) {
        const int buf = kt & 1;
        if (kt + 1 < KCHUNKS) {
            issue(kt + 1, buf ^ 1);
            __builtin_amdgcn_s_wait_asynccnt(12);   // drain chunk kt (12 async instr/wave)
        } else {
            __builtin_amdgcn_s_wait_asynccnt(0);
        }
        __syncthreads();

        const float*    Ab = (const float*)(Abase + buf * ABYTES);
        const _Float16* Bb = (const _Float16*)(Bbase + buf * BBYTES);

#pragma unroll
        for (int ks = 0; ks < 2; ++ks) {
            v16h a[2];
#pragma unroll
            for (int sm = 0; sm < 2; ++sm) {
                const float4* ap = (const float4*)(Ab + (mrow0 + sm * 16) * SA + ks * 32 + koff);
#pragma unroll
                for (int e = 0; e < 4; ++e) {
                    float4 f = ap[e];
                    a[sm][4 * e + 0] = (_Float16)f.x;
                    a[sm][4 * e + 1] = (_Float16)f.y;
                    a[sm][4 * e + 2] = (_Float16)f.z;
                    a[sm][4 * e + 3] = (_Float16)f.w;
                }
            }
#pragma unroll
            for (int sn = 0; sn < 4; ++sn) {
                const v8h* bp = (const v8h*)(Bb + (nrow0 + sn * 16) * SB + ks * 32 + (koff >> 1));
                v8h lo = bp[0], hi = bp[2];
                v16h b;
#pragma unroll
                for (int e = 0; e < 8; ++e) { b[e] = lo[e]; b[e + 8] = hi[e]; }
                acc[0][sn] = __builtin_amdgcn_wmma_f32_16x16x32_f16(
                    false, a[0], false, b, (short)0, acc[0][sn], false, false);
                acc[1][sn] = __builtin_amdgcn_wmma_f32_16x16x32_f16(
                    false, a[1], false, b, (short)0, acc[1][sn], false, false);
            }
        }
        __syncthreads();
    }

    const int mwbase = Mbase + mw * 32;
    const int mhalf  = (lane >> 4) << 3;
    const int ncol   = Nbase + nh * 64 + (lane & 15);
#pragma unroll
    for (int sm = 0; sm < 2; ++sm)
#pragma unroll
        for (int sn = 0; sn < 4; ++sn) {
            const int n_out = ncol + sn * 16;
#pragma unroll
            for (int r = 0; r < 8; ++r)
                out[(size_t)(mwbase + sm * 16 + mhalf + r) * 256 + n_out] =
                    acc[sm][sn][r] + bcol[sn];
        }
}
#endif  // HAVE_ASYNC

// ================= synchronous fallback (round-1 kernel) =================
#define LDS_STRIDE 88
__global__ __launch_bounds__(256) void binconv_wmma_f16(
    const float* __restrict__ x, const float* __restrict__ wk,
    const float* __restrict__ bias, float* __restrict__ out)
{
    __shared__ _Float16 Ash[BM * LDS_STRIDE];
    __shared__ _Float16 Bsh[BN * LDS_STRIDE];

    const int tid  = threadIdx.x;
    const int lane = tid & 31;
    const int w    = tid >> 5;
    const int mw   = w & 3;
    const int nh   = w >> 2;
    const int Mbase = blockIdx.x * BM;
    const int Nbase = blockIdx.y * BN;
    const int a_m  = tid & 127;
    const int a_kb = (tid >> 7) << 5;
    const int b_n  = tid & 127;
    const int b_kb = (tid >> 7) << 5;
    const int mg  = Mbase + a_m;
    const int img = mg / 12100;
    const int rem = mg - img * 12100;
    const int oy  = rem / 110;
    const int ox  = rem - oy * 110;

    float bcol[4];
#pragma unroll
    for (int sn = 0; sn < 4; ++sn)
        bcol[sn] = bias[Nbase + nh * 64 + sn * 16 + (lane & 15)];

    v8f acc[2][4];
#pragma unroll
    for (int i = 0; i < 2; ++i)
#pragma unroll
        for (int j = 0; j < 4; ++j) acc[i][j] = (v8f){};

    const int koffA = (lane >> 4) << 3;
    const int koffB = (lane >> 4) << 3;
    const int mrow0 = mw * 32 + (lane & 15);
    const int nrow0 = nh * 64 + (lane & 15);

    for (int kt = 0; kt < KCHUNKS; ++kt) {
        const int plane = kt >> 1;
        const int c0    = (kt & 1) << 6;
        const int ky    = plane / 3;
        const int kx    = plane - ky * 3;
        {
            const float4* s4 = (const float4*)(x +
                ((((img * 112) + (oy + ky)) * 112 + (ox + kx)) * 128 + c0 + a_kb));
#pragma unroll
            for (int i = 0; i < 4; ++i) {
                float4 f0 = s4[2 * i], f1 = s4[2 * i + 1];
                v8h h;
                h[0] = (_Float16)f0.x; h[1] = (_Float16)f0.y;
                h[2] = (_Float16)f0.z; h[3] = (_Float16)f0.w;
                h[4] = (_Float16)f1.x; h[5] = (_Float16)f1.y;
                h[6] = (_Float16)f1.z; h[7] = (_Float16)f1.w;
                *(v8h*)&Ash[a_m * LDS_STRIDE + a_kb + 8 * i] = h;
            }
        }
        {
            const int k0 = kt * BK;
            const float* wsrc = wk + (size_t)(k0 + b_kb) * 256 + Nbase + b_n;
#pragma unroll
            for (int i = 0; i < 4; ++i) {
                v8h h;
#pragma unroll
                for (int j = 0; j < 8; ++j) {
                    float v = wsrc[(i * 8 + j) * 256];
                    h[j] = (v > 0.0f) ? (_Float16)1.0f : (_Float16)-1.0f;
                }
                *(v8h*)&Bsh[b_n * LDS_STRIDE + b_kb + 8 * i] = h;
            }
        }
        __syncthreads();
#pragma unroll
        for (int ks = 0; ks < 2; ++ks) {
            const int kk = ks << 5;
            v16h a[2];
#pragma unroll
            for (int sm = 0; sm < 2; ++sm) {
                const _Float16* p = &Ash[(mrow0 + sm * 16) * LDS_STRIDE + kk + koffA];
                v8h lo = *(const v8h*)p;
                v8h hi = *(const v8h*)(p + 16);
#pragma unroll
                for (int e = 0; e < 8; ++e) { a[sm][e] = lo[e]; a[sm][e + 8] = hi[e]; }
            }
#pragma unroll
            for (int sn = 0; sn < 4; ++sn) {
                const _Float16* q = &Bsh[(nrow0 + sn * 16) * LDS_STRIDE + kk + koffB];
                v8h lo = *(const v8h*)q;
                v8h hi = *(const v8h*)(q + 16);
                v16h b;
#pragma unroll
                for (int e = 0; e < 8; ++e) { b[e] = lo[e]; b[e + 8] = hi[e]; }
                acc[0][sn] = __builtin_amdgcn_wmma_f32_16x16x32_f16(
                    false, a[0], false, b, (short)0, acc[0][sn], false, false);
                acc[1][sn] = __builtin_amdgcn_wmma_f32_16x16x32_f16(
                    false, a[1], false, b, (short)0, acc[1][sn], false, false);
            }
        }
        __syncthreads();
    }

    const int mwbase = Mbase + mw * 32;
    const int mhalf  = (lane >> 4) << 3;
    const int ncol   = Nbase + nh * 64 + (lane & 15);
#pragma unroll
    for (int sm = 0; sm < 2; ++sm)
#pragma unroll
        for (int sn = 0; sn < 4; ++sn) {
            const int n_out = ncol + sn * 16;
#pragma unroll
            for (int r = 0; r < 8; ++r)
                out[(size_t)(mwbase + sm * 16 + mhalf + r) * 256 + n_out] =
                    acc[sm][sn][r] + bcol[sn];
        }
}

extern "C" void kernel_launch(void* const* d_in, const int* in_sizes, int n_in,
                              void* d_out, int out_size, void* d_ws, size_t ws_size,
                              hipStream_t stream) {
    (void)in_sizes; (void)n_in; (void)out_size;
    const float* x    = (const float*)d_in[0];
    const float* wk   = (const float*)d_in[1];
    const float* bias = (const float*)d_in[2];
    float* out        = (float*)d_out;

    dim3 grid(3025, 2, 1);
    dim3 block(256, 1, 1);

#if HAVE_ASYNC
    const size_t wsB_bytes = (size_t)256 * 1152 * sizeof(_Float16);  // 589824
    if (ws_size >= wsB_bytes) {
        _Float16* wsB = (_Float16*)d_ws;
        binarize_B<<<dim3(1152, 1, 1), block, 0, stream>>>(wk, wsB);
        binconv_wmma_async<<<grid, block, SMEM_TOTAL, stream>>>(x, wsB, bias, out);
        return;
    }
#endif
    (void)d_ws; (void)ws_size;
    binconv_wmma_f16<<<grid, block, 0, stream>>>(x, wk, bias, out);
}
